// MultiNatten_38302518346173
// MI455X (gfx1250) — compile-verified, matrix operand order checked by reading, not run on hardware
//
#include <hip/hip_runtime.h>
#include <hip/hip_bf16.h>

typedef float v8f  __attribute__((ext_vector_type(8)));
typedef __bf16 v8bf  __attribute__((ext_vector_type(8)));
typedef __bf16 v16bf __attribute__((ext_vector_type(16)));

__device__ __forceinline__ float gelu_exact(float x) {
    return 0.5f * x * (1.0f + erff(x * 0.70710678118654752f));
}

// ---------------------------------------------------------------------------
// WMMA GEMM:  C[M,N] = A[M,K] * B[K,N]   (B optionally stored [N,K] when BT)
// block = 128 threads = 4 waves. Macro tile: 128 rows x 64 cols.
// Each wave: 32x64 (2 row-tiles x 4 col-tiles = 8 independent accumulators
// -> 8 back-to-back v_wmma per K-step, A frags reused x4, B frags x2).
// Double-buffered LDS staging (fp32 -> bf16), K-step = 32.
// EPI: 0 = BN+GELU (p0=g,p1=b,p2=mean,p3=var), 1 = +bias(p0),
//      2 = gelu(+bias), 3 = +bias + residual.  Batched over blockIdx.z.
// ---------------------------------------------------------------------------
template<bool BT>
__device__ __forceinline__ void stage_tiles(
    __bf16 (*__restrict__ lAb)[32], __bf16 (*__restrict__ lBb)[32],
    const float* __restrict__ A, int lda,
    const float* __restrict__ B, int ldb,
    int row0, int col0, int k0, int tid)
{
    // A: 128 rows x 32 k ; thread tid stages full row `tid`
    {
        const float* ap = A + (long)(row0 + tid) * lda + k0;
        #pragma unroll
        for (int i = 0; i < 32; i += 4) {
            float4 v4 = *(const float4*)(ap + i);
            lAb[tid][i + 0] = (__bf16)v4.x;
            lAb[tid][i + 1] = (__bf16)v4.y;
            lAb[tid][i + 2] = (__bf16)v4.z;
            lAb[tid][i + 3] = (__bf16)v4.w;
        }
    }
    // B: 64 n-rows x 32 k (stored transposed [n][k]); thread covers
    // n = tid>>1, k range [(tid&1)*16, +16)
    const int nn = tid >> 1;
    const int kh = (tid & 1) * 16;
    if constexpr (BT) {            // B stored [N,K] row-major
        const float* bp = B + (long)(col0 + nn) * ldb + k0 + kh;
        #pragma unroll
        for (int i = 0; i < 16; i += 4) {
            float4 v4 = *(const float4*)(bp + i);
            lBb[nn][kh + i + 0] = (__bf16)v4.x;
            lBb[nn][kh + i + 1] = (__bf16)v4.y;
            lBb[nn][kh + i + 2] = (__bf16)v4.z;
            lBb[nn][kh + i + 3] = (__bf16)v4.w;
        }
    } else {                       // B stored [K,N] row-major
        #pragma unroll
        for (int i = 0; i < 16; ++i)
            lBb[nn][kh + i] = (__bf16)B[(long)(k0 + kh + i) * ldb + col0 + nn];
    }
}

template<int EPI, bool BT>
__global__ __launch_bounds__(128)
void gemm_wmma_k(const float* __restrict__ A, int lda, long aBatch,
                 const float* __restrict__ B, int ldb, long bBatch,
                 float* __restrict__ C, int ldc, long cBatch,
                 int M, int N, int Kd,
                 const float* __restrict__ p0, long p0Batch,
                 const float* __restrict__ p1, const float* __restrict__ p2,
                 const float* __restrict__ p3,
                 const float* __restrict__ res, long resBatch, int resLd)
{
    __shared__ __bf16 lA[2][128][32];   // [m][k], double buffered
    __shared__ __bf16 lB[2][64][32];    // [n][k] (transposed tile)

    const int tid  = threadIdx.x;
    const int w    = tid >> 5;
    const int lane = tid & 31;
    const int half = lane >> 4;
    const int l16  = lane & 15;
    const int g    = blockIdx.z;

    A += (long)g * aBatch;
    B += (long)g * bBatch;
    C += (long)g * cBatch;
    if (p0)  p0  += (long)g * p0Batch;
    if (res) res += (long)g * resBatch;

    const int row0 = blockIdx.y * 128;
    const int col0 = blockIdx.x * 64;

    v8f acc[8];
    #pragma unroll
    for (int i = 0; i < 8; ++i) acc[i] = {};

    const int nsteps = Kd >> 5;
    stage_tiles<BT>(lA[0], lB[0], A, lda, B, ldb, row0, col0, 0, tid);
    __syncthreads();

    for (int s = 0; s < nsteps; ++s) {
        const int cur = s & 1;
        if (s + 2 < nsteps)   // L2 prefetch two K-steps ahead (global_prefetch_b8)
            __builtin_prefetch(A + (long)(row0 + tid) * lda + (s + 2) * 32, 0, 1);
        if (s + 1 < nsteps)
            stage_tiles<BT>(lA[cur ^ 1], lB[cur ^ 1], A, lda, B, ldb,
                            row0, col0, (s + 1) * 32, tid);

        // A fragments for this wave's two 16-row tiles
        v16bf af[2];
        #pragma unroll
        for (int rt = 0; rt < 2; ++rt) {
            const int m = w * 32 + rt * 16 + l16;
            v8bf alo = *(const v8bf*)&lA[cur][m][8 * half];
            v8bf ahi = *(const v8bf*)&lA[cur][m][16 + 8 * half];
            af[rt] = __builtin_shufflevector(alo, ahi,
                         0,1,2,3,4,5,6,7,8,9,10,11,12,13,14,15);
        }
        // 4 col tiles x 2 row tiles = 8 independent WMMAs
        #pragma unroll
        for (int nt = 0; nt < 4; ++nt) {
            v8bf blo = *(const v8bf*)&lB[cur][nt * 16 + l16][8 * half];
            v8bf bhi = *(const v8bf*)&lB[cur][nt * 16 + l16][16 + 8 * half];
            v16bf bf = __builtin_shufflevector(blo, bhi,
                           0,1,2,3,4,5,6,7,8,9,10,11,12,13,14,15);
            #pragma unroll
            for (int rt = 0; rt < 2; ++rt) {
                acc[rt * 4 + nt] = __builtin_amdgcn_wmma_f32_16x16x32_bf16(
                    false, af[rt], false, bf, (short)0, acc[rt * 4 + nt],
                    false, false);
            }
        }
        __syncthreads();
    }

    // ---- epilogue + store (C layout: VGPR r -> row r + 8*half, lane -> col)
    #pragma unroll
    for (int rt = 0; rt < 2; ++rt) {
        #pragma unroll
        for (int nt = 0; nt < 4; ++nt) {
            const int col = col0 + nt * 16 + l16;
            #pragma unroll
            for (int r = 0; r < 8; ++r) {
                const int row = row0 + w * 32 + rt * 16 + r + 8 * half;
                float v = acc[rt * 4 + nt][r];
                if constexpr (EPI == 0) {
                    v = (v - p2[col]) * rsqrtf(p3[col] + 1e-5f) * p0[col] + p1[col];
                    v = gelu_exact(v);
                } else if constexpr (EPI == 1) {
                    v += p0[col];
                } else if constexpr (EPI == 2) {
                    v = gelu_exact(v + p0[col]);
                } else if constexpr (EPI == 3) {
                    v += p0[col] + res[(long)row * resLd + col];
                }
                C[(long)row * ldc + col] = v;
            }
        }
    }
}

// ---------------------------------------------------------------------------
// Group LayerNorm over C=64 channels; one thread per (group, token).
// addr = in + g*gStride + t*inLd + c
// ---------------------------------------------------------------------------
__global__ __launch_bounds__(256)
void ln_group_k(const float* __restrict__ in, int inLd, long gStride,
                const float* __restrict__ gamma, const float* __restrict__ beta,
                float* __restrict__ outp)
{
    const int t = blockIdx.x * 256 + threadIdx.x;
    const int g = blockIdx.y;
    const float* xp = in + (long)g * gStride + (long)t * inLd;
    float v[64];
    float s = 0.f;
    #pragma unroll
    for (int c = 0; c < 64; ++c) { v[c] = xp[c]; s += v[c]; }
    const float mean = s * (1.f / 64.f);
    float var = 0.f;
    #pragma unroll
    for (int c = 0; c < 64; ++c) { float d = v[c] - mean; var += d * d; }
    const float rs = rsqrtf(var * (1.f / 64.f) + 1e-5f);
    float* op = outp + ((long)g * 4096 + t) * 64;
    #pragma unroll
    for (int c = 0; c < 64; ++c)
        op[c] = (v[c] - mean) * rs * gamma[g * 64 + c] + beta[g * 64 + c];
}

// ---------------------------------------------------------------------------
// 7x7 neighborhood attention, flash-style online softmax.
// One thread per (group, token, head);  d = 32, heads = 2, K = 7, H = W = 32.
// qkv layout per token: [q(2x32) | k(2x32) | v(2x32)] = 192 floats.
// ---------------------------------------------------------------------------
__global__ __launch_bounds__(256)
void natten_k(const float* __restrict__ qkv, const float* __restrict__ rpb,
              float* __restrict__ outp)
{
    const int idx  = blockIdx.x * 256 + threadIdx.x;   // 0..8191
    const int g    = blockIdx.y;
    const int head = idx >> 12;
    const int t    = idx & 4095;
    const int b    = t >> 10;
    const int i    = (t >> 5) & 31;
    const int j    = t & 31;

    const float scale = 0.1767766952966369f;           // 32^-0.5
    const float* qp = qkv + ((long)g * 4096 + t) * 192 + head * 32;
    float q[32], o[32];
    #pragma unroll
    for (int d = 0; d < 32; ++d) { q[d] = qp[d] * scale; o[d] = 0.f; }

    int si = i - 3; si = si < 0 ? 0 : (si > 25 ? 25 : si);
    int sj = j - 3; sj = sj < 0 ? 0 : (sj > 25 ? 25 : sj);
    const float* rp = rpb + (long)(g * 2 + head) * 169;   // 13*13

    float m = -1e30f, l = 0.f;
    for (int u = 0; u < 7; ++u) {
        for (int v = 0; v < 7; ++v) {
            const int nt = (b << 10) + ((si + u) << 5) + (sj + v);
            const float* kp = qkv + ((long)g * 4096 + nt) * 192 + 64 + head * 32;
            const float* vp = kp + 64;
            float s = 0.f;
            #pragma unroll
            for (int d = 0; d < 32; ++d) s += q[d] * kp[d];
            s += rp[(si + u - i + 6) * 13 + (sj + v - j + 6)];
            const float mn  = fmaxf(m, s);
            const float cor = __expf(m - mn);
            const float p   = __expf(s - mn);
            l = l * cor + p;
            #pragma unroll
            for (int d = 0; d < 32; ++d) o[d] = o[d] * cor + p * vp[d];
            m = mn;
        }
    }
    const float inv = 1.0f / l;
    float* op = outp + ((long)g * 4096 + t) * 64 + head * 32;
    #pragma unroll
    for (int d = 0; d < 32; ++d) op[d] = o[d] * inv;
}

// ---------------------------------------------------------------------------
// Global average pool over H*W=1024 positions. One thread per (b, ct).
// ---------------------------------------------------------------------------
__global__ __launch_bounds__(256)
void pool_k(const float* __restrict__ outs, float* __restrict__ pooled)
{
    const int idx = blockIdx.x * 256 + threadIdx.x;    // B*512 = 2048
    const int b = idx >> 9, ct = idx & 511;
    const int g = ct >> 6,  c  = ct & 63;
    const float* p = outs + ((long)g * 4096 + b * 1024) * 64 + c;
    float s = 0.f;
    for (int pos = 0; pos < 1024; ++pos) s += p[pos * 64];
    pooled[idx] = s * (1.f / 1024.f);
}

// gates[g,b,c] = sigmoid( sum_t pooled[b,t]*fc_w[g,t,c] + fc_b[g,c] )
__global__ __launch_bounds__(256)
void gates_k(const float* __restrict__ pooled, const float* __restrict__ fc_w,
             const float* __restrict__ fc_b, float* __restrict__ gates)
{
    const int idx = blockIdx.x * 256 + threadIdx.x;    // G*B*C = 2048
    const int g = idx >> 8;
    const int b = (idx >> 6) & 3;
    const int c = idx & 63;
    float s = fc_b[g * 64 + c];
    const float* pw = pooled + b * 512;
    const float* ww = fc_w + (long)g * 512 * 64 + c;
    for (int t = 0; t < 512; ++t) s += pw[t] * ww[t * 64];
    gates[(g * 4 + b) * 64 + c] = 1.f / (1.f + __expf(-s));
}

// ---------------------------------------------------------------------------
// Final: gate groups, LayerNorm over 512 channels. One wave32 per token,
// each lane owns 16 contiguous channels; wave shuffle reduction.
// ---------------------------------------------------------------------------
__global__ __launch_bounds__(256)
void final_ln_k(const float* __restrict__ outs, const float* __restrict__ gates,
                const float* __restrict__ lng, const float* __restrict__ lnb,
                float* __restrict__ out)
{
    const int w    = threadIdx.x >> 5;
    const int lane = threadIdx.x & 31;
    const int t    = blockIdx.x * 8 + w;
    const int b    = t >> 10;
    const int g    = lane >> 2;              // 16 chans stay inside one group
    const int c0   = (lane & 3) * 16;

    const float* op = outs  + ((long)g * 4096 + t) * 64 + c0;
    const float* gp = gates + (g * 4 + b) * 64 + c0;
    float vals[16], s = 0.f, ss = 0.f;
    #pragma unroll
    for (int e = 0; e < 16; ++e) {
        float x = op[e] * gp[e];
        vals[e] = x; s += x; ss += x * x;
    }
    #pragma unroll
    for (int off = 16; off >= 1; off >>= 1) {
        s  += __shfl_xor(s,  off, 32);
        ss += __shfl_xor(ss, off, 32);
    }
    const float mean = s * (1.f / 512.f);
    const float var  = ss * (1.f / 512.f) - mean * mean;
    const float rs   = rsqrtf(var + 1e-5f);
    const int ct0 = lane * 16;
    float* dst = out + (long)t * 512 + ct0;
    #pragma unroll
    for (int e = 0; e < 16; ++e)
        dst[e] = (vals[e] - mean) * rs * lng[ct0 + e] + lnb[ct0 + e];
}

// ---------------------------------------------------------------------------
extern "C" void kernel_launch(void* const* d_in, const int* in_sizes, int n_in,
                              void* d_out, int out_size, void* d_ws, size_t ws_size,
                              hipStream_t stream)
{
    const float* x      = (const float*)d_in[0];
    const float* conv_w = (const float*)d_in[1];
    const float* bn_g   = (const float*)d_in[2];
    const float* bn_b   = (const float*)d_in[3];
    const float* bn_m   = (const float*)d_in[4];
    const float* bn_v   = (const float*)d_in[5];
    const float* n1_g   = (const float*)d_in[6];
    const float* n1_b   = (const float*)d_in[7];
    const float* qkv_w  = (const float*)d_in[8];
    const float* qkv_b  = (const float*)d_in[9];
    const float* rpb    = (const float*)d_in[10];
    const float* proj_w = (const float*)d_in[11];
    const float* proj_b = (const float*)d_in[12];
    const float* n2_g   = (const float*)d_in[13];
    const float* n2_b   = (const float*)d_in[14];
    const float* fc1_w  = (const float*)d_in[15];
    const float* fc1_b  = (const float*)d_in[16];
    const float* fc2_w  = (const float*)d_in[17];
    const float* fc2_b  = (const float*)d_in[18];
    const float* fc_w   = (const float*)d_in[19];
    const float* fc_b   = (const float*)d_in[20];
    const float* ln_g   = (const float*)d_in[21];
    const float* ln_b   = (const float*)d_in[22];
    float* out = (float*)d_out;
    float* ws  = (float*)d_ws;

    const long T = 4096;                         // B*H*W tokens
    float* w_y    = ws;                          // T*512
    float* w_ln   = w_y   + T * 512;             // G*T*64  (ln1 out, then attn out)
    float* w_qkv  = w_ln  + 8 * T * 64;          // G*T*192 (qkv, then ln2 out)
    float* w_yres = w_qkv + 8 * T * 192;         // G*T*64
    float* w_hid  = w_yres + 8 * T * 64;         // G*T*256
    float* w_outs = w_hid  + 8 * T * 256;        // G*T*64
    float* w_pool = w_outs + 8 * T * 64;         // B*512
    float* w_gate = w_pool + 4 * 512;            // G*B*64
    float* w_attn = w_ln;                        // reuse: free after qkv GEMM
    float* w_ln2  = w_qkv;                       // reuse: free after attention

    // 1) stem: y = gelu(bn(x @ conv_w^T))   [4096,512] = [4096,256]x[256,512]
    gemm_wmma_k<0, true><<<dim3(8, 32, 1), 128, 0, stream>>>(
        x, 256, 0,  conv_w, 256, 0,  w_y, 512, 0,  4096, 512, 256,
        bn_g, 0, bn_b, bn_m, bn_v,  nullptr, 0, 0);

    // 2) LN1 per group (input: column slice of y)
    ln_group_k<<<dim3(16, 8), 256, 0, stream>>>(w_y, 512, 64, n1_g, n1_b, w_ln);

    // 3) qkv: [4096,192] = [4096,64]x[64,192], batched over 8 groups
    gemm_wmma_k<1, false><<<dim3(3, 32, 8), 128, 0, stream>>>(
        w_ln, 64, T * 64,  qkv_w, 192, 64 * 192,  w_qkv, 192, T * 192,
        4096, 192, 64,  qkv_b, 192, nullptr, nullptr, nullptr,  nullptr, 0, 0);

    // 4) neighborhood attention
    natten_k<<<dim3(32, 8), 256, 0, stream>>>(w_qkv, rpb, w_attn);

    // 5) proj + residual(y group slice): [4096,64]x[64,64]
    gemm_wmma_k<3, false><<<dim3(1, 32, 8), 128, 0, stream>>>(
        w_attn, 64, T * 64,  proj_w, 64, 64 * 64,  w_yres, 64, T * 64,
        4096, 64, 64,  proj_b, 64, nullptr, nullptr, nullptr,  w_y, 64, 512);

    // 6) LN2 per group
    ln_group_k<<<dim3(16, 8), 256, 0, stream>>>(w_yres, 64, T * 64, n2_g, n2_b, w_ln2);

    // 7) fc1 + gelu: [4096,256] = [4096,64]x[64,256]
    gemm_wmma_k<2, false><<<dim3(4, 32, 8), 128, 0, stream>>>(
        w_ln2, 64, T * 64,  fc1_w, 256, 64 * 256,  w_hid, 256, T * 256,
        4096, 256, 64,  fc1_b, 256, nullptr, nullptr, nullptr,  nullptr, 0, 0);

    // 8) fc2 + residual(yres): [4096,64] = [4096,256]x[256,64]
    gemm_wmma_k<3, false><<<dim3(1, 32, 8), 128, 0, stream>>>(
        w_hid, 256, T * 256,  fc2_w, 64, 256 * 64,  w_outs, 64, T * 64,
        4096, 64, 256,  fc2_b, 64, nullptr, nullptr, nullptr,  w_yres, T * 64, 64);

    // 9) global pool, 10) gates, 11) gated final LN
    pool_k<<<dim3(8), 256, 0, stream>>>(w_outs, w_pool);
    gates_k<<<dim3(8), 256, 0, stream>>>(w_pool, fc_w, fc_b, w_gate);
    final_ln_k<<<dim3(512), 256, 0, stream>>>(w_outs, w_gate, ln_g, ln_b, out);
}
